// LearnedStructuredAttentionSB_25563645346310
// MI455X (gfx1250) — compile-verified, matrix operand order checked by reading, not run on hardware
//
#include <hip/hip_runtime.h>
#include <hip/hip_bf16.h>

// ---------------------------------------------------------------------------
// LearnedStructuredAttention (S=2048, B=4, H=1024) for MI455X / gfx1250.
// Matrix-core bound (~120 GFLOP vs ~140MB HBM). All GEMMs on
// v_wmma_f32_16x16x32_f16 with 2x4 register blocking per wave; tiles staged
// via CDNA5 async global->LDS (ASYNCcnt) with double buffering and a peeled
// final iteration (branch-free steady-state mainloop).
// ---------------------------------------------------------------------------

typedef __attribute__((ext_vector_type(16))) _Float16 v16h;
typedef __attribute__((ext_vector_type(4)))  _Float16 h4;
typedef __attribute__((ext_vector_type(8)))  float    v8f;

constexpr int S_ = 2048;
constexpr int B_ = 4;
constexpr int H_ = 1024;

constexpr int BM = 128;  // M tile (4 waves x 32 rows)
constexpr int BN = 64;   // N tile
constexpr int BK = 32;   // K tile (= WMMA K)
// 80B LDS rows: 16B-aligned (async B128 stores) and 20-bank stride whose
// cycle mod 64 banks covers 16 distinct banks -> conflict-free frag reads.
constexpr int LDA = 40;

// ---------------------------------------------------------------------------
// CDNA5 async global->LDS copy (16B/lane, ASYNCcnt) + waits.
// ---------------------------------------------------------------------------
__device__ inline void async_ld16(const void* g, unsigned lds_byte_off) {
  asm volatile("global_load_async_to_lds_b128 %0, %1, off"
               :: "v"(lds_byte_off), "v"(g)
               : "memory");
}
__device__ inline void wait_async_0() {
  asm volatile("s_wait_asynccnt 0x0" ::: "memory");
}
__device__ inline void wait_async_le6() {  // oldest 6 (current tile) done
  asm volatile("s_wait_asynccnt 0x6" ::: "memory");
}

// ---------------------------------------------------------------------------
// Fragment loader (ISA 7.12.2, wave32): lane holds row = lane&15; 16 halfs
// cover K={0..7,16..23} (lanes 0-15) / K={8..15,24..31} (lanes 16-31).
// Lowers to 2x ds_load_b128 per fragment.
// ---------------------------------------------------------------------------
__device__ inline v16h load_frag(const _Float16* __restrict__ p, int lane) {
  const _Float16* row = p + (lane & 15) * LDA;
  const int khi = (lane & 16) >> 1;
  v16h f;
#pragma unroll
  for (int r = 0; r < 8; ++r) {
    const int kb = ((r < 4) ? 2 * r : 16 + 2 * (r - 4)) + khi;
    f[2 * r]     = row[kb];
    f[2 * r + 1] = row[kb + 1];
  }
  return f;
}

#define WMMA_F16(a, b, c) \
  __builtin_amdgcn_wmma_f32_16x16x32_f16(false, (a), false, (b), (short)0, (c), false, false)

// ---------------------------------------------------------------------------
// Shared double-buffered GEMM mainloop.  C[128x64] per block, A,B row-major
// f16 with row stride == K (true for all three GEMMs here).
// LDS layout: [buf0: A(128xLDA) B(64xLDA)][buf1: same].
// ---------------------------------------------------------------------------
constexpr int LDS_HALFS = 2 * (BM + BN) * LDA;  // 30720 B

template <int KD>
__device__ inline void gemm_mainloop(const _Float16* __restrict__ A,
                                     const _Float16* __restrict__ Bm,
                                     int m0, int n0, int t,
                                     _Float16* lds, v8f (&acc)[2][4]) {
  const int lane = t & 31, wave = t >> 5;
  const int r0 = t >> 2, col = t & 3;  // one 16B chunk per 32-row group
  const unsigned base = (unsigned)(unsigned long long)(uintptr_t)lds;
  constexpr unsigned TOG = (unsigned)((BM + BN) * LDA * 2);  // buffer toggle

  // Per-thread staging addresses (bumped by +BK per k-step).
  const _Float16* ga[4];
  unsigned la[4];
#pragma unroll
  for (int i = 0; i < 4; ++i) {
    ga[i] = A + (size_t)(m0 + r0 + 32 * i) * KD + col * 8;
    la[i] = base + (unsigned)((r0 + 32 * i) * (LDA * 2) + col * 16);
  }
  const _Float16* gb[2];
  unsigned lb[2];
#pragma unroll
  for (int i = 0; i < 2; ++i) {
    gb[i] = Bm + (size_t)(n0 + r0 + 32 * i) * KD + col * 8;
    lb[i] = base + (unsigned)(BM * LDA * 2 + (r0 + 32 * i) * (LDA * 2) + col * 16);
  }

  // Prologue: stage k-tile 0 into buffer 0 (6 async ops per wave).
#pragma unroll
  for (int i = 0; i < 4; ++i) { async_ld16(ga[i], la[i]); ga[i] += BK; }
#pragma unroll
  for (int i = 0; i < 2; ++i) { async_ld16(gb[i], lb[i]); gb[i] += BK; }

  // Steady state: branch-free; always stages the next tile.
  unsigned ib = 0;
  for (int k0 = 0; k0 < KD - BK; k0 += BK) {
    const unsigned off = (ib ^ 1u) * TOG;
#pragma unroll
    for (int i = 0; i < 4; ++i) { async_ld16(ga[i], la[i] + off); ga[i] += BK; }
#pragma unroll
    for (int i = 0; i < 2; ++i) { async_ld16(gb[i], lb[i] + off); gb[i] += BK; }
    wait_async_le6();  // in-order completion: current tile has landed
    __syncthreads();

    const _Float16* pa = lds + ib * ((BM + BN) * LDA);
    const _Float16* pb = pa + BM * LDA;
    const v16h a0 = load_frag(pa + (32 * wave) * LDA, lane);
    const v16h a1 = load_frag(pa + (32 * wave + 16) * LDA, lane);
#pragma unroll
    for (int nb = 0; nb < 4; ++nb) {
      const v16h b = load_frag(pb + 16 * nb * LDA, lane);
      acc[0][nb] = WMMA_F16(a0, b, acc[0][nb]);
      acc[1][nb] = WMMA_F16(a1, b, acc[1][nb]);
    }
    __syncthreads();  // reads done before next iter's stage overwrites
    ib ^= 1u;
  }

  // Peeled final tile: nothing left to stage.
  wait_async_0();
  __syncthreads();
  {
    const _Float16* pa = lds + ib * ((BM + BN) * LDA);
    const _Float16* pb = pa + BM * LDA;
    const v16h a0 = load_frag(pa + (32 * wave) * LDA, lane);
    const v16h a1 = load_frag(pa + (32 * wave + 16) * LDA, lane);
#pragma unroll
    for (int nb = 0; nb < 4; ++nb) {
      const v16h b = load_frag(pb + 16 * nb * LDA, lane);
      acc[0][nb] = WMMA_F16(a0, b, acc[0][nb]);
      acc[1][nb] = WMMA_F16(a1, b, acc[1][nb]);
    }
  }
}

// ---------------------------------------------------------------------------
// Kernel 0: vectorized fp32 -> f16 conversion (float4 -> half4).
// ---------------------------------------------------------------------------
__global__ __launch_bounds__(256)
void cvt_f16_kernel(const float4* __restrict__ in, h4* __restrict__ out, int n4) {
  const int i = blockIdx.x * 256 + threadIdx.x;
  if (i < n4) {
    const float4 v = in[i];
    h4 o;
    o[0] = (_Float16)v.x; o[1] = (_Float16)v.y;
    o[2] = (_Float16)v.z; o[3] = (_Float16)v.w;
    out[i] = o;
  }
}

// ---------------------------------------------------------------------------
// Kernel 1: QKV projection GEMM (all-f16 sources).  Q[m,o]=sum_h x[m,h]W[o,h].
// Q,K stored [B][S][H]; V stored TRANSPOSED [B][H][S].
// ---------------------------------------------------------------------------
__global__ __launch_bounds__(128)
void qkv_gemm_kernel(const _Float16* __restrict__ xh,
                     const _Float16* __restrict__ wh,  // [3][H][H]
                     const float* __restrict__ bq, const float* __restrict__ bk,
                     const float* __restrict__ bv,
                     _Float16* __restrict__ qh, _Float16* __restrict__ kh,
                     _Float16* __restrict__ vt) {
  const int which = blockIdx.z;
  const _Float16* __restrict__ W = wh + (size_t)which * H_ * H_;
  const float* __restrict__ bias = (which == 0) ? bq : (which == 1) ? bk : bv;

  __shared__ __align__(16) _Float16 lds[LDS_HALFS];
  const int m0 = blockIdx.x * BM;  // over S*B = 8192
  const int n0 = blockIdx.y * BN;  // over H
  const int t = threadIdx.x;
  const int lane = t & 31, wave = t >> 5;

  v8f acc[2][4] = {};
  gemm_mainloop<H_>(xh, W, m0, n0, t, lds, acc);

  if (which < 2) {
    _Float16* __restrict__ outp = which ? kh : qh;
#pragma unroll
    for (int mf = 0; mf < 2; ++mf)
#pragma unroll
      for (int nb = 0; nb < 4; ++nb) {
        const int n = n0 + 16 * nb + (lane & 15);
        const float bv_ = bias[n];
#pragma unroll
        for (int r = 0; r < 8; ++r) {
          const int m = m0 + 32 * wave + 16 * mf + r + ((lane >> 4) << 3);
          outp[(size_t)(m & 3) * S_ * H_ + (size_t)(m >> 2) * H_ + n] =
              (_Float16)(acc[mf][nb][r] + bv_);
        }
      }
  } else {
#pragma unroll
    for (int mf = 0; mf < 2; ++mf)
#pragma unroll
      for (int nb = 0; nb < 4; ++nb) {
        const int n = n0 + 16 * nb + (lane & 15);
        const float bv_ = bias[n];
#pragma unroll
        for (int r = 0; r < 8; ++r) {
          const int m = m0 + 32 * wave + 16 * mf + r + ((lane >> 4) << 3);
          vt[(size_t)(m & 3) * H_ * S_ + (size_t)n * S_ + (m >> 2)] =
              (_Float16)(acc[mf][nb][r] + bv_);
        }
      }
  }
}

// ---------------------------------------------------------------------------
// Kernel 2: scores[b,q,k] = (Q.K)/32 + beta*ssm
// ---------------------------------------------------------------------------
__global__ __launch_bounds__(128)
void scores_kernel(const _Float16* __restrict__ qh,
                   const _Float16* __restrict__ kh,
                   const float* __restrict__ ssm,
                   const float* __restrict__ betaPtr,
                   float* __restrict__ sc) {
  const int bb = blockIdx.z;
  const float beta = betaPtr[0];
  const float inv_scale = 0.03125f;  // 1/sqrt(1024)

  __shared__ __align__(16) _Float16 lds[LDS_HALFS];
  const int m0 = blockIdx.x * BM;
  const int n0 = blockIdx.y * BN;
  const int t = threadIdx.x;
  const int lane = t & 31, wave = t >> 5;

  v8f acc[2][4] = {};
  gemm_mainloop<H_>(qh + (size_t)bb * S_ * H_, kh + (size_t)bb * S_ * H_,
                    m0, n0, t, lds, acc);

#pragma unroll
  for (int mf = 0; mf < 2; ++mf)
#pragma unroll
    for (int nb = 0; nb < 4; ++nb) {
      const int n = n0 + 16 * nb + (lane & 15);
#pragma unroll
      for (int r = 0; r < 8; ++r) {
        const int m = m0 + 32 * wave + 16 * mf + r + ((lane >> 4) << 3);
        const size_t off = (size_t)bb * S_ * S_ + (size_t)m * S_ + n;
        sc[off] = acc[mf][nb][r] * inv_scale + beta * ssm[off];
      }
    }
}

// ---------------------------------------------------------------------------
// Kernel 3: row softmax over 2048 fp32 -> f16 probabilities.
// ---------------------------------------------------------------------------
__global__ __launch_bounds__(256)
void softmax_kernel(const float* __restrict__ sc, _Float16* __restrict__ P) {
  const size_t row = blockIdx.x;
  const float* __restrict__ src = sc + row * (size_t)S_;
  _Float16* __restrict__ dst = P + row * (size_t)S_;
  const int t = threadIdx.x;

  float v[8];
#pragma unroll
  for (int i = 0; i < 8; ++i) v[i] = src[t + 256 * i];

  float mx = v[0];
#pragma unroll
  for (int i = 1; i < 8; ++i) mx = fmaxf(mx, v[i]);
#pragma unroll
  for (int off = 16; off > 0; off >>= 1) mx = fmaxf(mx, __shfl_xor(mx, off, 32));

  __shared__ float red[8];
  if ((t & 31) == 0) red[t >> 5] = mx;
  __syncthreads();
  float m2 = red[0];
#pragma unroll
  for (int i = 1; i < 8; ++i) m2 = fmaxf(m2, red[i]);

  float sum = 0.f;
#pragma unroll
  for (int i = 0; i < 8; ++i) {
    v[i] = __expf(v[i] - m2);
    sum += v[i];
  }
#pragma unroll
  for (int off = 16; off > 0; off >>= 1) sum += __shfl_xor(sum, off, 32);
  __syncthreads();
  if ((t & 31) == 0) red[t >> 5] = sum;
  __syncthreads();
  float s2 = 0.f;
#pragma unroll
  for (int i = 0; i < 8; ++i) s2 += red[i];
  const float inv = 1.0f / s2;

#pragma unroll
  for (int i = 0; i < 8; ++i) dst[t + 256 * i] = (_Float16)(v[i] * inv);
}

// ---------------------------------------------------------------------------
// Kernel 4: out[q,b,h] = sum_k P[b,q,k] * Vt[b,h,k]
// ---------------------------------------------------------------------------
__global__ __launch_bounds__(128)
void out_gemm_kernel(const _Float16* __restrict__ P,
                     const _Float16* __restrict__ vt,
                     float* __restrict__ out) {
  const int bb = blockIdx.z;

  __shared__ __align__(16) _Float16 lds[LDS_HALFS];
  const int m0 = blockIdx.x * BM;  // query rows
  const int n0 = blockIdx.y * BN;  // H columns
  const int t = threadIdx.x;
  const int lane = t & 31, wave = t >> 5;

  v8f acc[2][4] = {};
  gemm_mainloop<S_>(P + (size_t)bb * S_ * S_, vt + (size_t)bb * H_ * S_,
                    m0, n0, t, lds, acc);

#pragma unroll
  for (int mf = 0; mf < 2; ++mf)
#pragma unroll
    for (int nb = 0; nb < 4; ++nb) {
      const int n = n0 + 16 * nb + (lane & 15);
#pragma unroll
      for (int r = 0; r < 8; ++r) {
        const int m = m0 + 32 * wave + 16 * mf + r + ((lane >> 4) << 3);
        out[(size_t)m * (B_ * H_) + (size_t)bb * H_ + n] = acc[mf][nb][r];
      }
    }
}

// ---------------------------------------------------------------------------
extern "C" void kernel_launch(void* const* d_in, const int* in_sizes, int n_in,
                              void* d_out, int out_size, void* d_ws,
                              size_t ws_size, hipStream_t stream) {
  const float* x    = (const float*)d_in[0];
  const float* ssm  = (const float*)d_in[1];
  const float* Wq   = (const float*)d_in[2];
  const float* bq   = (const float*)d_in[3];
  const float* Wk   = (const float*)d_in[4];
  const float* bk   = (const float*)d_in[5];
  const float* Wv   = (const float*)d_in[6];
  const float* bv   = (const float*)d_in[7];
  const float* beta = (const float*)d_in[8];
  float* out = (float*)d_out;

  // ws: xh | wh[3] | qh | kh | vt (f16) | scores (f32) | P (f16)  ~166 MB
  const size_t BSH = (size_t)B_ * S_ * H_;  // 8M
  const size_t HH  = (size_t)H_ * H_;       // 1M
  const size_t BSS = (size_t)B_ * S_ * S_;  // 16M
  _Float16* xh = (_Float16*)d_ws;
  _Float16* wh = xh + BSH;       // [3][H][H]
  _Float16* qh = wh + 3 * HH;
  _Float16* kh = qh + BSH;
  _Float16* vt = kh + BSH;
  float*    sc = (float*)(vt + BSH);
  _Float16* P  = (_Float16*)(sc + BSS);

  // 0) fp32 -> f16 conversions (vectorized)
  cvt_f16_kernel<<<dim3((int)(BSH / 4 / 256)), 256, 0, stream>>>(
      (const float4*)x, (h4*)xh, (int)(BSH / 4));
  cvt_f16_kernel<<<dim3((int)(HH / 4 / 256)), 256, 0, stream>>>(
      (const float4*)Wq, (h4*)(wh + 0 * HH), (int)(HH / 4));
  cvt_f16_kernel<<<dim3((int)(HH / 4 / 256)), 256, 0, stream>>>(
      (const float4*)Wk, (h4*)(wh + 1 * HH), (int)(HH / 4));
  cvt_f16_kernel<<<dim3((int)(HH / 4 / 256)), 256, 0, stream>>>(
      (const float4*)Wv, (h4*)(wh + 2 * HH), (int)(HH / 4));

  // 1) QKV projections (z selects Q/K/V)
  qkv_gemm_kernel<<<dim3((S_ * B_) / BM, H_ / BN, 3), 128, 0, stream>>>(
      xh, wh, bq, bk, bv, qh, kh, vt);

  // 2) scores = QK^T/sqrt(H) + beta*ssm
  scores_kernel<<<dim3(S_ / BM, S_ / BN, B_), 128, 0, stream>>>(
      qh, kh, ssm, beta, sc);

  // 3) softmax
  softmax_kernel<<<dim3(B_ * S_), 256, 0, stream>>>(sc, P);

  // 4) out = P V
  out_gemm_kernel<<<dim3(S_ / BM, H_ / BN, B_), 128, 0, stream>>>(P, vt, out);
}
